// Net_62216896250024
// MI455X (gfx1250) — matrix operand order repaired and checked by reading, not compiled, not run on hardware
//
#include <hip/hip_runtime.h>
#include <hip/hip_bf16.h>
#include <cmath>
#include <stdint.h>

// MI455X / gfx1250, wave32. Heavy GEMMs via v_wmma_f32_16x16x32_bf16 with
// double-buffered LDS filled by global_load_async_to_lds_b128 (ASYNCcnt),
// hardware v_tanh_f32 epilogues, K as a compile-time constant so the
// pair-select in the pipelined K loop is a shift, not a division.

typedef __bf16 bf16;
typedef __attribute__((ext_vector_type(16))) __bf16 v16bf;
typedef __attribute__((ext_vector_type(8)))  __bf16 v8bf;
typedef __attribute__((ext_vector_type(8)))  float  v8f;

#define B_  64
#define L_  32
#define V_  32000
#define E_  512
#define H_  1024
#define MROWS 5952   /* 3*64*31 */

// ---------------------------------------------------------------- asm helpers
__device__ __forceinline__ void async_cp_b128(unsigned ldsOff, uint64_t gaddr) {
    // per-lane 16B copy: LDS[ldsOff] = MEM[gaddr]; tracked by ASYNCcnt
    asm volatile("global_load_async_to_lds_b128 %0, %1, off"
                 :: "v"(ldsOff), "v"(gaddr) : "memory");
}
__device__ __forceinline__ void async_wait0() {
    asm volatile("s_wait_asynccnt 0x0" ::: "memory");
}
__device__ __forceinline__ float fast_tanh(float x) {
    float r;
    asm("v_tanh_f32 %0, %1" : "=v"(r) : "v"(x));
    return r;
}

// ---------------------------------------------------------------- conversions
__global__ void cvt_f32_bf16(const float* __restrict__ in, bf16* __restrict__ out, long n) {
    long i = (long)blockIdx.x * blockDim.x + threadIdx.x;
    long stride = (long)gridDim.x * blockDim.x;
    for (; i < n; i += stride) out[i] = (bf16)in[i];
}

__global__ void zero_bf16(bf16* __restrict__ p, long n) {
    long i = (long)blockIdx.x * blockDim.x + threadIdx.x;
    long stride = (long)gridDim.x * blockDim.x;
    for (; i < n; i += stride) p[i] = (bf16)0.0f;
}

// ------------------------------------------------- embedding gather + mean pool
__global__ void embed_pool(const int* __restrict__ x, const float* __restrict__ emb,
                           bf16* __restrict__ pooled) {
    int blk = blockIdx.x;            // 0..191 = s*64 + b
    int s = blk >> 6, b = blk & 63;
    __shared__ int toks[L_];
    if (threadIdx.x < L_) toks[threadIdx.x] = x[b * 4 * L_ + s * L_ + threadIdx.x];
    __syncthreads();
    for (int e = threadIdx.x; e < E_; e += blockDim.x) {
        float acc = 0.0f;
        #pragma unroll 4
        for (int l = 0; l < L_; ++l) acc += emb[(long)toks[l] * E_ + e];
        pooled[(long)blk * E_ + e] = (bf16)(acc * (1.0f / L_));
    }
}

// ---------------------------------------------------------- word index gather
__global__ void make_word_idx(const int* __restrict__ x, int* __restrict__ widx) {
    int r = blockIdx.x * blockDim.x + threadIdx.x;
    if (r < MROWS) {
        int s = r / 1984, rem = r % 1984, j = rem / 64, b = rem % 64;
        widx[r] = x[b * 4 * L_ + (s + 1) * L_ + j];
    }
}

// --------------------------------------------------------- one-hot first rows
__global__ void first_rows(const int* __restrict__ x, float* __restrict__ out) {
    int blk = blockIdx.x;            // 0..191
    int s = blk / 64, b = blk % 64;
    long row = (long)(b * 96 + s * 32) * V_;
    int tok = x[b * 4 * L_ + (s + 1) * L_];
    for (int v = threadIdx.x; v < V_; v += blockDim.x)
        out[row + v] = (v == tok) ? 1.0f : 0.0f;
}

// --------------------------------------------------------------- WMMA GEMM
// C[M,N] = act( sum_p A_p[M,KT] @ W_p[KT,N] (+ bias) )
// 256 threads = 8 waves. Block tile 64(M) x 128(N). Wave (wm=wave>>2, wn=wave&3)
// computes 32x32 = 2x2 WMMA tiles -> 4 v_wmma per K-step per wave.
// LDS double buffer, filled with global_load_async_to_lds_b128.
#define TM 64
#define TN 128
#define ABUF (TM * 32)          /* bf16 elems */
#define BBUF (32 * TN)
#define BUFSZ (ABUF + BBUF)

template <int KT, int NPAIR, bool TANH, int BIAS, int ROWMAP, int BATCH>
__global__ __launch_bounds__(256) void wmma_gemm(
    const bf16* __restrict__ A0, const bf16* __restrict__ W0,
    const bf16* __restrict__ A1, const bf16* __restrict__ W1,
    int M, int N,
    float* __restrict__ Cf, bf16* __restrict__ Cb,
    const int* __restrict__ wordIdx, const float* __restrict__ WwF) {

    __shared__ __align__(16) bf16 smem[2 * BUFSZ];

    const int z    = blockIdx.z;
    const int m0   = blockIdx.y * TM;
    const int n0   = blockIdx.x * TN;
    const int tid  = threadIdx.x;
    const int lane = tid & 31;
    const int wave = tid >> 5;
    const int wm = wave >> 2;          // 0..1 -> M offset wm*32
    const int wn = wave & 3;           // 0..3 -> N offset wn*32

    const bf16* Ap[2];
    const bf16* Wp[2];
    if (BATCH == 1) {
        Ap[0] = A0 + (long)(z / 31) * 64 * KT;
        Wp[0] = W0 + (long)((z % 31) + 1) * KT * N;
    } else {
        Ap[0] = A0; Wp[0] = W0;
    }
    Ap[1] = A1; Wp[1] = W1;

    const unsigned sbase = (unsigned)(uintptr_t)(&smem[0]);   // LDS byte offset
    constexpr int kpb   = KT >> 5;           // 32-wide K blocks per pair (pow2)
    constexpr int iters = NPAIR * kpb;

    // stage K-block `it` into LDS buffer `buf` via async copies
    auto stage = [&](int it, int buf) {
        int p  = it / kpb;                   // compile-time pow2 -> shift
        int kb = (it & (kpb - 1)) << 5;
        const bf16* gA = Ap[p];
        const bf16* gW = Wp[p];
        unsigned aoff = sbase + (unsigned)buf * (BUFSZ * 2);
        unsigned boff = aoff + ABUF * 2;
        {   // A tile 64x32: 256 x 16B
            int r = tid >> 2, seg = tid & 3;
            async_cp_b128(aoff + r * 64 + seg * 16,
                          (uint64_t)(gA + (long)(m0 + r) * KT + kb + seg * 8));
        }
        {   // B tile 32x128: 512 x 16B
            #pragma unroll
            for (int c = 0; c < 2; ++c) {
                int q = c * 256 + tid, kr = q >> 4, seg = q & 15;
                async_cp_b128(boff + kr * 256 + seg * 16,
                              (uint64_t)(gW + (long)(kb + kr) * N + n0 + seg * 8));
            }
        }
        if (kb + 32 < KT)   // speculative prefetch of the tile after next
            __builtin_prefetch(gW + (long)(kb + 32 + (tid >> 4)) * N + n0, 0, 0);
    };

    v8f accs[4] = {};
    const int rbase = wm * 32 + (lane & 15);       // A lane -> M row (tile 0)
    const int k0l   = (lane & 16) ? 8 : 0;         // A lane -> K half select

    stage(0, 0);
    async_wait0();
    __syncthreads();

    int cur = 0;
    #pragma unroll 1
    for (int it = 0; it < iters; ++it) {
        if (it + 1 < iters) stage(it + 1, cur ^ 1);

        const bf16* LA = smem + cur * BUFSZ;
        const bf16* LB = LA + ABUF;

        union { v16bf v; v8bf h[2]; } a0, a1;
        a0.h[0] = *(const v8bf*)&LA[rbase * 32 + k0l];
        a0.h[1] = *(const v8bf*)&LA[rbase * 32 + k0l + 16];
        a1.h[0] = *(const v8bf*)&LA[(rbase + 16) * 32 + k0l];
        a1.h[1] = *(const v8bf*)&LA[(rbase + 16) * 32 + k0l + 16];
        // B operand: lane = N column (lane%16), 16 K values in table order (same as A)
        v16bf b0, b1;
        #pragma unroll
        for (int s = 0; s < 16; ++s) {
            int kk = (s < 8) ? (k0l + s) : (16 + k0l + s - 8);
            b0[s] = LB[kk * TN + wn * 32 + (lane & 15)];
            b1[s] = LB[kk * TN + wn * 32 + 16 + (lane & 15)];
        }

        accs[0] = __builtin_amdgcn_wmma_f32_16x16x32_bf16(
            false, a0.v, false, b0, (short)0, accs[0], false, false);
        accs[1] = __builtin_amdgcn_wmma_f32_16x16x32_bf16(
            false, a0.v, false, b1, (short)0, accs[1], false, false);
        accs[2] = __builtin_amdgcn_wmma_f32_16x16x32_bf16(
            false, a1.v, false, b0, (short)0, accs[2], false, false);
        accs[3] = __builtin_amdgcn_wmma_f32_16x16x32_bf16(
            false, a1.v, false, b1, (short)0, accs[3], false, false);

        async_wait0();
        __syncthreads();
        cur ^= 1;
    }

    // epilogue: C/D layout -> VGPR i: lanes 0-15 M=i, lanes 16-31 M=i+8; N=lane%16
    bf16* cb = (Cb && BATCH == 1) ? (Cb + (long)z * 64 * N) : Cb;
    #pragma unroll
    for (int mi = 0; mi < 2; ++mi) {
        #pragma unroll
        for (int ni = 0; ni < 2; ++ni) {
            v8f acc = accs[mi * 2 + ni];
            int ncol = n0 + wn * 32 + ni * 16 + (lane & 15);
            #pragma unroll
            for (int i = 0; i < 8; ++i) {
                int m = m0 + wm * 32 + mi * 16 + i + ((lane & 16) ? 8 : 0);
                float v = acc[i];
                if (BIAS == 1) {
                    int word = wordIdx[m];
                    v += WwF[(long)word * H_ + ncol];
                }
                if (TANH) v = fast_tanh(v);
                if (Cf) {
                    long orow;
                    if (ROWMAP == 1) {
                        int s = m / 1984, rem = m % 1984, j = rem / 64, b = rem % 64;
                        orow = (long)(b * 96 + s * 32 + j + 1);
                    } else {
                        orow = m;
                    }
                    Cf[orow * (long)N + ncol] = v;
                }
                if (Cb) cb[(long)m * N + ncol] = (bf16)v;
            }
        }
    }
}

// ------------------------------------------------------------------- launcher
extern "C" void kernel_launch(void* const* d_in, const int* in_sizes, int n_in,
                              void* d_out, int out_size, void* d_ws, size_t ws_size,
                              hipStream_t stream) {
    (void)in_sizes; (void)n_in; (void)out_size; (void)ws_size;
    const int*   x    = (const int*)d_in[0];
    const float* emb  = (const float*)d_in[2];
    const float* Wcsm = (const float*)d_in[3];
    const float* Wx1  = (const float*)d_in[4];
    const float* Wh1  = (const float*)d_in[5];
    const float* Wx2  = (const float*)d_in[6];
    const float* Wh2  = (const float*)d_in[7];
    const float* U    = (const float*)d_in[8];
    const float* Ww   = (const float*)d_in[9];
    const float* Wc   = (const float*)d_in[10];
    const float* Wfc  = (const float*)d_in[11];
    float* out = (float*)d_out;

    char* w = (char*)d_ws;
    auto alloc = [&](size_t bytes) -> char* {
        char* p = w; w += (bytes + 255) & ~(size_t)255; return p;
    };
    bf16* Wcsm_b = (bf16*)alloc((size_t)E_ * H_ * 2);
    bf16* Wx1_b  = (bf16*)alloc((size_t)H_ * H_ * 2);
    bf16* Wh1_b  = (bf16*)alloc((size_t)H_ * H_ * 2);
    bf16* Wx2_b  = (bf16*)alloc((size_t)H_ * H_ * 2);
    bf16* Wh2_b  = (bf16*)alloc((size_t)H_ * H_ * 2);
    bf16* U_b    = (bf16*)alloc((size_t)L_ * H_ * H_ * 2);
    bf16* Wc_b   = (bf16*)alloc((size_t)H_ * H_ * 2);
    bf16* Wfc_b  = (bf16*)alloc((size_t)H_ * V_ * 2);
    bf16* pooled = (bf16*)alloc((size_t)192 * E_ * 2);
    bf16* sent   = (bf16*)alloc((size_t)192 * H_ * 2);
    bf16* zerobf = (bf16*)alloc((size_t)64 * H_ * 2);
    bf16* h1s    = (bf16*)alloc((size_t)3 * 64 * H_ * 2);
    bf16* hs     = (bf16*)alloc((size_t)3 * 64 * H_ * 2);
    bf16* ctx    = (bf16*)alloc((size_t)MROWS * H_ * 2);
    bf16* cur    = (bf16*)alloc((size_t)MROWS * H_ * 2);
    int*  widx   = (int*)alloc((size_t)MROWS * 4);

    const int CT = 256;
    auto cvt = [&](const float* src, bf16* dst, long n) {
        int blocks = (int)((n + CT * 8 - 1) / (CT * 8));
        if (blocks > 4096) blocks = 4096;
        cvt_f32_bf16<<<blocks, CT, 0, stream>>>(src, dst, n);
    };
    cvt(Wcsm, Wcsm_b, (long)E_ * H_);
    cvt(Wx1,  Wx1_b,  (long)H_ * H_);
    cvt(Wh1,  Wh1_b,  (long)H_ * H_);
    cvt(Wx2,  Wx2_b,  (long)H_ * H_);
    cvt(Wh2,  Wh2_b,  (long)H_ * H_);
    cvt(U,    U_b,    (long)L_ * H_ * H_);
    cvt(Wc,   Wc_b,   (long)H_ * H_);
    cvt(Wfc,  Wfc_b,  (long)H_ * V_);
    zero_bf16<<<64, CT, 0, stream>>>(zerobf, (long)64 * H_);

    embed_pool<<<192, CT, 0, stream>>>(x, emb, pooled);
    make_word_idx<<<(MROWS + CT - 1) / CT, CT, 0, stream>>>(x, widx);

    // CSM: sent = tanh(pooled @ Wcsm)   [192,512]x[512,1024]
    wmma_gemm<E_, 1, true, 0, 0, 0><<<dim3(H_ / TN, 192 / TM, 1), CT, 0, stream>>>(
        pooled, Wcsm_b, nullptr, nullptr, 192, H_,
        nullptr, sent, nullptr, nullptr);

    // RCM: 3 sequential steps x 2 fused-GEMM layers
    for (int t = 0; t < 3; ++t) {
        const bf16* h1p = (t == 0) ? zerobf : (h1s + (t - 1) * 64 * H_);
        const bf16* h2p = (t == 0) ? zerobf : (hs  + (t - 1) * 64 * H_);
        wmma_gemm<H_, 2, true, 0, 0, 0><<<dim3(H_ / TN, 1, 1), CT, 0, stream>>>(
            sent + (size_t)t * 64 * H_, Wx1_b, h1p, Wh1_b, 64, H_,
            nullptr, h1s + (size_t)t * 64 * H_, nullptr, nullptr);
        wmma_gemm<H_, 2, true, 0, 0, 0><<<dim3(H_ / TN, 1, 1), CT, 0, stream>>>(
            h1s + (size_t)t * 64 * H_, Wx2_b, h2p, Wh2_b, 64, H_,
            nullptr, hs + (size_t)t * 64 * H_, nullptr, nullptr);
    }

    // contexts: ctx[(s*31+j)*64 + b] = tanh(hs[s] @ U[j+1]); 93 batched GEMMs
    wmma_gemm<H_, 1, true, 0, 0, 1><<<dim3(H_ / TN, 1, 93), CT, 0, stream>>>(
        hs, U_b, nullptr, nullptr, 64, H_,
        nullptr, ctx, nullptr, nullptr);

    // cur = tanh(Ww[idx] + ctx @ Wc)
    wmma_gemm<H_, 1, true, 1, 0, 0><<<dim3(H_ / TN, MROWS / TM, 1), CT, 0, stream>>>(
        ctx, Wc_b, nullptr, nullptr, MROWS, H_,
        nullptr, cur, widx, Ww);

    // y_sec = cur @ Wfc  scattered into out rows b*96 + s*32 + j + 1
    wmma_gemm<H_, 1, false, 0, 1, 0><<<dim3(V_ / TN, MROWS / TM, 1), CT, 0, stream>>>(
        cur, Wfc_b, nullptr, nullptr, MROWS, V_,
        out, nullptr, nullptr, nullptr);

    // one-hot rows at l = 0 of each section
    first_rows<<<192, CT, 0, stream>>>(x, out);
}
